// ModuleRelProp2_24790551232691
// MI455X (gfx1250) — matrix operand order repaired and checked by reading, not compiled, not run on hardware
//
#include <hip/hip_runtime.h>
#include <hip/hip_bf16.h>

// ---- problem constants (from reference) ----
#define B_  2
#define K_  384
#define N_  4096
#define D_  1024
#define H_  150
#define HP_ 160      // H padded to multiple of 32 for WMMA K-dim
#define L_  12
#define LP_ 16       // L padded to 16 for WMMA N-dim
#define NB_ 10
#define DD_ 20
#define BK_ (B_*K_)  // 768

typedef __attribute__((ext_vector_type(16))) _Float16 v16h;
typedef __attribute__((ext_vector_type(2)))  _Float16 v2h;
typedef __attribute__((ext_vector_type(8)))  float    v8f;
typedef long long i64;

enum { ACT_NONE = 0, ACT_TANH = 1, ACT_SIGMOID = 2 };

__device__ __forceinline__ v2h pk_f16(float a, float b)
{
    return __builtin_bit_cast(v2h, __builtin_amdgcn_cvt_pkrtz(a, b));
}

// Fragment-order index helpers (16-bit WMMA operand register layouts):
// A 16x32 (m,k):  lane = (m&15) + ((k>>3)&1)*16 ; pos = (k>>4)*8 + (k&7)
// B 32x16 (k,n):  lane = (n&15) + ((k>>3)&1)*16 ; pos = (k>>4)*8 + (k&7)
// For an even/odd k pair, lane is identical and positions are consecutive,
// so a packed-f16 pair can be stored with one aligned b32 LDS store.

// =====================================================================
// Generic batched GEMM: C[z][M][Nstore] = act((A[z] @ W[z] + bias) * rowscale)
// Requirements (guaranteed by callers): M % 64 == 0, Kd % 32 == 0.
// f32 operands converted to f16 (packed pairs) and stored to LDS in
// *fragment order*; each lane loads its whole v16h with 2x ds_load_b128.
// Block = 4 waves, 64x64 tile; each wave owns a 16x64 strip.
// =====================================================================
__global__ void __launch_bounds__(128)
gemm_wmma_kernel(const float* __restrict__ A, const float* __restrict__ W,
                 const float* __restrict__ bias, const float* __restrict__ rowscale,
                 float* __restrict__ C,
                 int M, int Kd, int Nvalid, int Nstore,
                 i64 sA, i64 sW, i64 sC, i64 sRS, int act)
{
    __shared__ v16h AsF[4][32];   // [wave][lane] A fragments (4 KB)
    __shared__ v16h WsF[4][32];   // [n-subtile][lane] B fragments (4 KB)

    const int tid  = threadIdx.x;
    const int lane = tid & 31;
    const int wv   = tid >> 5;
    const int hi   = lane >> 4;
    const int lm   = lane & 15;
    const int z    = blockIdx.z;
    const i64 aBase = (i64)z * sA, wBase = (i64)z * sW, cBase = (i64)z * sC;
    const int m0 = blockIdx.x * 64;
    const int n0 = blockIdx.y * 64;

    v2h* asp2 = (v2h*)AsF;
    v2h* wsp2 = (v2h*)WsF;

    v8f zero8 = {0.f,0.f,0.f,0.f,0.f,0.f,0.f,0.f};
    v8f acc[4];
    acc[0] = zero8; acc[1] = zero8; acc[2] = zero8; acc[3] = zero8;

    for (int k0 = 0; k0 < Kd; k0 += 32) {
        // stage A 64x32: each thread handles 8 even-k pairs (b64 global loads)
        for (int e = tid; e < 64 * 16; e += 128) {
            int r  = e >> 4;                 // tile row 0..63
            int kp = (e & 15) << 1;          // even k 0..30
            const float* ap = &A[aBase + (i64)(m0 + r) * Kd + (k0 + kp)];
            v2h pk = pk_f16(ap[0], ap[1]);
            int dl = (r & 15) + (((kp >> 3) & 1) << 4);
            int dp = ((kp >> 4) << 3) + (kp & 7);        // even
            asp2[(((r >> 4) * 32) + dl) * 8 + (dp >> 1)] = pk;
        }
        // stage W 32x64: each thread handles 8 (even-k pair, n) elements;
        // branch-free tail masking for Nvalid (clamp + multiply)
        for (int e = tid; e < 16 * 64; e += 128) {
            int kp = (e >> 6) << 1;          // even k 0..30
            int c  = e & 63;                 // n within tile
            int gn  = n0 + c;
            int gnc = gn < Nvalid ? gn : (Nvalid - 1);
            float msk = gn < Nvalid ? 1.f : 0.f;
            const float* wp = &W[wBase + (i64)(k0 + kp) * Nvalid + gnc];
            v2h pk = pk_f16(wp[0] * msk, wp[Nvalid] * msk);
            int dl = (c & 15) + (((kp >> 3) & 1) << 4);
            wsp2[(((c >> 4) * 32) + dl) * 8 + ((((kp >> 4) << 3) + (kp & 7)) >> 1)] = pk;
        }
        __syncthreads();

        v16h fa = AsF[wv][lane];                    // 2x ds_load_b128
        #pragma unroll
        for (int t = 0; t < 4; ++t) {
            v16h fb = WsF[t][lane];
            acc[t] = __builtin_amdgcn_wmma_f32_16x16x32_f16(
                false, fa, false, fb, (short)0, acc[t], false, false);
        }
        __syncthreads();
    }

    // epilogue: C layout -> VGPR cv holds row (cv + hi*8), col = lane%16
    #pragma unroll
    for (int t = 0; t < 4; ++t) {
        #pragma unroll
        for (int cv = 0; cv < 8; ++cv) {
            int m = m0 + wv * 16 + hi * 8 + cv;
            int n = n0 + t * 16 + lm;
            if (n < Nstore) {
                float vv = 0.f;
                if (n < Nvalid) {
                    vv = acc[t][cv];
                    if (bias)     vv += bias[n];
                    if (rowscale) vv *= rowscale[sRS * z + m];
                    if (act == ACT_TANH)         vv = tanhf(vv);
                    else if (act == ACT_SIGMOID) vv = 1.f / (1.f + __expf(-vv));
                }
                C[cBase + (i64)m * Nstore + n] = vv;
            }
        }
    }
}

// =====================================================================
// dH[nb][HP_] = dist_emb[nb] @ Wd + bd   (padded cols -> 0)
// =====================================================================
__global__ void dh_kernel(const float* __restrict__ dist_emb,
                          const float* __restrict__ Wd,
                          const float* __restrict__ bd,
                          float* __restrict__ dH)
{
    int nb = blockIdx.x;      // 0..NB_-1
    int h  = threadIdx.x;     // 0..HP_-1
    float v = 0.f;
    if (h < H_) {
        v = bd[h];
        for (int dd = 0; dd < DD_; ++dd)
            v += dist_emb[nb * DD_ + dd] * Wd[dd * H_ + h];
    }
    dH[nb * HP_ + h] = v;
}

// Pack Wo [H_,L_] -> f16 *fragment-order* WoPF[5][32][16]:
//   k-step s, lane, pos  ->  k = s*32 + (lane>>4)*16 + pos, n = lane&15
// bo -> padded boP[LP_]
__global__ void pack_wo_kernel(const float* __restrict__ Wo, const float* __restrict__ bo,
                               _Float16* __restrict__ WoPF, float* __restrict__ boP)
{
    int idx = blockIdx.x * 256 + threadIdx.x;
    if (idx < 5 * 32 * 16) {
        int s    = idx >> 9;            // /512
        int rem  = idx & 511;
        int lane = rem >> 4;
        int pos  = rem & 15;
        int n    = lane & 15;
        int k    = s * 32 + ((lane >> 4) << 4) + pos;
        WoPF[idx] = (_Float16)((k < H_ && n < L_) ? Wo[k * L_ + n] : 0.f);
    }
    if (idx < LP_) boP[idx] = (idx < L_) ? bo[idx] : 0.f;
}

__device__ __forceinline__ int bucketf(int d)
{
    int da = d < 0 ? -d : d;
    if (da <= 4) return da;
    int lg = 31 - __clz(da);          // floor(log2(da)), da >= 5 here
    int b  = lg + 3;
    return b < (NB_ - 1) ? b : (NB_ - 1);
}

// =====================================================================
// Fused pairwise scorer:
//   scores[b,i,j,:] (+)= relu(l_i + r_j + dH[bucket(begin_j-end_i)]) @ Wo + bo
// One block per (b,i). A-fragments built in registers (pk f32->f16),
// B-fragments are coalesced 32B loads from pre-swizzled WoPF.
// =====================================================================
__global__ void __launch_bounds__(128)
scorer_kernel(const float* __restrict__ lbuf, const float* __restrict__ rbuf,
              const float* __restrict__ dH,
              const _Float16* __restrict__ WoPF, const float* __restrict__ boP,
              const int* __restrict__ span_begin, const int* __restrict__ span_end,
              float* __restrict__ scores, int accumulate)
{
    __shared__ float l_s[HP_];
    __shared__ float dh_s[NB_][HP_];

    const int tid  = threadIdx.x;
    const int lane = tid & 31;
    const int wv   = tid >> 5;
    const int hi   = lane >> 4;
    const int lm   = lane & 15;
    const int bi   = blockIdx.x;      // 0..B*K-1
    const int b    = bi / K_;
    const int i    = bi - b * K_;

    for (int e = tid; e < HP_; e += 128) l_s[e] = lbuf[(i64)bi * HP_ + e];
    for (int e = tid; e < NB_ * HP_; e += 128) dh_s[e / HP_][e % HP_] = dH[e];
    __syncthreads();

    const int endi = span_end[b * K_ + i];

    // preload Wo B-fragments for all 5 k-steps (one 32B load each)
    const v16h* wof = (const v16h*)WoPF;
    v16h fb[5];
    #pragma unroll
    for (int s = 0; s < 5; ++s) fb[s] = wof[s * 32 + lane];

    const float bo_n = boP[lm];
    v8f cinit;
    #pragma unroll
    for (int q = 0; q < 8; ++q) cinit[q] = bo_n;

    for (int jt = wv; jt < K_ / 16; jt += 4) {   // uniform per wave
        const int j    = jt * 16 + lm;
        const int bkt  = bucketf(span_begin[b * K_ + j] - endi);
        const float* rrow = rbuf + ((i64)b * K_ + j) * HP_;

        v8f acc = cinit;
        #pragma unroll
        for (int s = 0; s < 5; ++s) {
            v16h fa;
            #pragma unroll
            for (int v = 0; v < 8; ++v) {
                int kk = s * 32 + (hi << 4) + (v << 1);
                float h0 = l_s[kk]     + rrow[kk]     + dh_s[bkt][kk];
                float h1 = l_s[kk + 1] + rrow[kk + 1] + dh_s[bkt][kk + 1];
                v2h pk = pk_f16(fmaxf(h0, 0.f), fmaxf(h1, 0.f));
                fa[2 * v]     = pk.x;
                fa[2 * v + 1] = pk.y;
            }
            acc = __builtin_amdgcn_wmma_f32_16x16x32_f16(
                false, fa, false, fb[s], (short)0, acc, false, false);
        }

        if (lm < L_) {
            #pragma unroll
            for (int cv = 0; cv < 8; ++cv) {
                int jr = jt * 16 + hi * 8 + cv;
                i64 idx = (((i64)b * K_ + i) * K_ + jr) * L_ + lm;
                float vv = acc[cv];
                if (accumulate) vv += scores[idx];
                scores[idx] = vv;
            }
        }
    }
}

// =====================================================================
// probs: P[b,i,j] = sigmoid(max_l scores) * mask; also PT (transpose) and
// rs[b,i] = 1/(rowsum+1e-7). Deterministic tree reduction (no atomics).
// =====================================================================
__global__ void __launch_bounds__(128)
probs_kernel(const float* __restrict__ scores, const float* __restrict__ mask,
             float* __restrict__ P, float* __restrict__ PT, float* __restrict__ rs)
{
    __shared__ float red[128];
    const int bi = blockIdx.x;
    const int b  = bi / K_;
    const int i  = bi - b * K_;
    float sum = 0.f;
    for (int j = threadIdx.x; j < K_; j += 128) {
        const float* sp = scores + (((i64)b * K_ + i) * K_ + j) * L_;
        float mx = sp[0];
        #pragma unroll
        for (int l = 1; l < L_; ++l) mx = fmaxf(mx, sp[l]);
        float p = (1.f / (1.f + __expf(-mx))) * mask[((i64)b * K_ + i) * K_ + j];
        P [((i64)b * K_ + i) * K_ + j] = p;
        PT[((i64)b * K_ + j) * K_ + i] = p;
        sum += p;
    }
    red[threadIdx.x] = sum;
    __syncthreads();
    for (int s = 64; s > 0; s >>= 1) {
        if (threadIdx.x < s) red[threadIdx.x] += red[threadIdx.x + s];
        __syncthreads();
    }
    if (threadIdx.x == 0) rs[bi] = 1.f / (red[0] + 1e-7f);
}

// out[row] = 1/(sum_j Pm[row][j] + 1e-7) over rows of a [B*K][K] matrix
__global__ void __launch_bounds__(128)
rowsum_inv_kernel(const float* __restrict__ Pm, float* __restrict__ out)
{
    __shared__ float red[128];
    const i64 base = (i64)blockIdx.x * K_;
    float s = 0.f;
    for (int j = threadIdx.x; j < K_; j += 128) s += Pm[base + j];
    red[threadIdx.x] = s;
    __syncthreads();
    for (int st = 64; st > 0; st >>= 1) {
        if (threadIdx.x < st) red[threadIdx.x] += red[threadIdx.x + st];
        __syncthreads();
    }
    if (threadIdx.x == 0) out[blockIdx.x] = 1.f / (red[0] + 1e-7f);
}

// concat [u | c1 | c2] -> cat3 [BK, 3D]
__global__ void cat3_kernel(const float* __restrict__ u, const float* __restrict__ c1,
                            const float* __restrict__ c2, float* __restrict__ cat)
{
    i64 idx = (i64)blockIdx.x * 256 + threadIdx.x;   // exact multiple
    int col = (int)(idx % (3 * D_));
    i64 row = idx / (3 * D_);
    float v;
    if      (col < D_)     v = u [row * D_ + col];
    else if (col < 2 * D_) v = c1[row * D_ + (col - D_)];
    else                   v = c2[row * D_ + (col - 2 * D_)];
    cat[idx] = v;
}

// concat [u | ctxt] -> cat2 [BK, 2D]
__global__ void cat2_kernel(const float* __restrict__ u, const float* __restrict__ c,
                            float* __restrict__ cat)
{
    i64 idx = (i64)blockIdx.x * 256 + threadIdx.x;
    int col = (int)(idx % (2 * D_));
    i64 row = idx / (2 * D_);
    cat[idx] = (col < D_) ? u[row * D_ + col] : c[row * D_ + (col - D_)];
}

// u_next = g*u + (1-g)*ctxt
__global__ void gate_kernel(const float* __restrict__ g, const float* __restrict__ u,
                            const float* __restrict__ c, float* __restrict__ un)
{
    i64 idx = (i64)blockIdx.x * 256 + threadIdx.x;
    float gg = g[idx];
    un[idx] = gg * u[idx] + (1.f - gg) * c[idx];
}

// overwrite_spans scatter: last valid duplicate wins (prune_indices sorted)
__global__ void scatter_kernel(const float* __restrict__ u, const int* __restrict__ prune,
                               const int* __restrict__ lens, float* __restrict__ out_update)
{
    int bk = blockIdx.x;
    int b = bk / K_, k = bk - b * K_;
    int len = lens[b];
    if (k >= len) return;
    int p = prune[b * K_ + k];
    if (k + 1 < len && prune[b * K_ + k + 1] == p) return;  // later dup wins
    const float* src = u + ((i64)b * K_ + k) * D_;
    float* dst = out_update + ((i64)b * N_ + p) * D_;
    for (int d = threadIdx.x; d < D_; d += blockDim.x) dst[d] = src[d];
}

// ---- host-side launch helper ----
static void gemm(const float* A, const float* W, const float* bias, const float* rsc,
                 float* C, int M, int Kd, int Nv, int Ns,
                 i64 sA, i64 sW, i64 sC, i64 sRS, int act, int batch, hipStream_t stream)
{
    dim3 grid((M + 63) / 64, (Ns + 63) / 64, batch);
    gemm_wmma_kernel<<<grid, 128, 0, stream>>>(A, W, bias, rsc, C, M, Kd, Nv, Ns,
                                               sA, sW, sC, sRS, act);
}

extern "C" void kernel_launch(void* const* d_in, const int* in_sizes, int n_in,
                              void* d_out, int out_size, void* d_ws, size_t ws_size,
                              hipStream_t stream)
{
    const float* span_vecs    = (const float*)d_in[0];
    const float* all_span     = (const float*)d_in[1];
    const int*   span_begin   = (const int*)d_in[2];
    const int*   span_end     = (const int*)d_in[3];
    const float* square_mask  = (const float*)d_in[4];
    const int*   prune        = (const int*)d_in[5];
    const int*   span_lengths = (const int*)d_in[6];
    const float* Wl  = (const float*)d_in[8];
    const float* bl  = (const float*)d_in[9];
    const float* Wr  = (const float*)d_in[10];
    const float* br  = (const float*)d_in[11];
    const float* dist_emb = (const float*)d_in[12];
    const float* Wd  = (const float*)d_in[13];
    const float* bd  = (const float*)d_in[14];
    const float* Wo  = (const float*)d_in[15];
    const float* bo  = (const float*)d_in[16];
    const float* Wff1 = (const float*)d_in[17];
    const float* Wff2 = (const float*)d_in[18];
    const float* Wg  = (const float*)d_in[19];
    const float* bg  = (const float*)d_in[20];

    float* out_update = (float*)d_out;
    float* out_u      = out_update + (size_t)B_ * N_ * D_;
    float* out_scores = out_u + (size_t)B_ * K_ * D_;

    // workspace carve (floats)
    float* w = (float*)d_ws;
    size_t off = 0;
    float* lbuf  = w + off; off += (size_t)BK_ * HP_;
    float* rbuf  = w + off; off += (size_t)BK_ * HP_;
    float* dHbuf = w + off; off += (size_t)NB_ * HP_;
    float* boP   = w + off; off += LP_;
    float* Pbuf  = w + off; off += (size_t)B_ * K_ * K_;
    float* PTbuf = w + off; off += (size_t)B_ * K_ * K_;
    float* rsbuf = w + off; off += BK_;
    float* csbuf = w + off; off += BK_;
    float* c1    = w + off; off += (size_t)BK_ * D_;      // ctxt1 then tanh-out
    float* c2    = w + off; off += (size_t)BK_ * D_;      // ctxt2 then ctxt
    float* gbuf  = w + off; off += (size_t)BK_ * D_;
    float* uA    = w + off; off += (size_t)BK_ * D_;
    float* uB    = w + off; off += (size_t)BK_ * D_;
    float* cat3  = w + off; off += (size_t)BK_ * 3 * D_;
    float* cat2  = w + off; off += (size_t)BK_ * 2 * D_;
    _Float16* WoPF = (_Float16*)(w + off); off += (5 * 32 * 16 + 1) / 2;

    // --- precompute folded tables ---
    pack_wo_kernel<<<(5 * 32 * 16 + 255) / 256, 256, 0, stream>>>(Wo, bo, WoPF, boP);
    dh_kernel<<<NB_, HP_, 0, stream>>>(dist_emb, Wd, bd, dHbuf);

    // --- initial projections + scorer (writes scores) ---
    gemm(span_vecs, Wl, bl, nullptr, lbuf, BK_, D_, H_, HP_, 0, 0, 0, 0, ACT_NONE, 1, stream);
    gemm(span_vecs, Wr, br, nullptr, rbuf, BK_, D_, H_, HP_, 0, 0, 0, 0, ACT_NONE, 1, stream);
    scorer_kernel<<<BK_, 128, 0, stream>>>(lbuf, rbuf, dHbuf, WoPF, boP,
                                           span_begin, span_end, out_scores, 0);

    const float* ucur = span_vecs;
    float* unext_arr[2] = { uA, uB };
    const i64 KK = (i64)K_ * K_, KD = (i64)K_ * D_;

    for (int it = 0; it < 2; ++it) {
        probs_kernel<<<BK_, 128, 0, stream>>>(out_scores, square_mask, Pbuf, PTbuf, rsbuf);
        rowsum_inv_kernel<<<BK_, 128, 0, stream>>>(PTbuf, csbuf);

        // ctxt1 = (P @ u) * rs ; ctxt2 = (P^T @ u) * cs   (batched over B)
        gemm(Pbuf,  ucur, nullptr, rsbuf, c1, K_, K_, D_, D_, KK, KD, KD, K_, ACT_NONE, B_, stream);
        gemm(PTbuf, ucur, nullptr, csbuf, c2, K_, K_, D_, D_, KK, KD, KD, K_, ACT_NONE, B_, stream);

        cat3_kernel<<<(BK_ * 3 * D_) / 256, 256, 0, stream>>>(ucur, c1, c2, cat3);
        gemm(cat3, Wff1, nullptr, nullptr, c1, BK_, 3 * D_, D_, D_, 0, 0, 0, 0, ACT_TANH, 1, stream);
        gemm(c1,   Wff2, nullptr, nullptr, c2, BK_, D_,     D_, D_, 0, 0, 0, 0, ACT_NONE, 1, stream);

        cat2_kernel<<<(BK_ * 2 * D_) / 256, 256, 0, stream>>>(ucur, c2, cat2);
        gemm(cat2, Wg, bg, nullptr, gbuf, BK_, 2 * D_, D_, D_, 0, 0, 0, 0, ACT_SIGMOID, 1, stream);

        float* unext = unext_arr[it];
        gate_kernel<<<(BK_ * D_) / 256, 256, 0, stream>>>(gbuf, ucur, c2, unext);

        gemm(unext, Wl, bl, nullptr, lbuf, BK_, D_, H_, HP_, 0, 0, 0, 0, ACT_NONE, 1, stream);
        gemm(unext, Wr, br, nullptr, rbuf, BK_, D_, H_, HP_, 0, 0, 0, 0, ACT_NONE, 1, stream);
        scorer_kernel<<<BK_, 128, 0, stream>>>(lbuf, rbuf, dHbuf, WoPF, boP,
                                               span_begin, span_end, out_scores, 1);
        ucur = unext;
    }

    // outputs: u, update_all (base copy + scatter)
    (void)hipMemcpyAsync(out_u, ucur, (size_t)BK_ * D_ * sizeof(float),
                         hipMemcpyDeviceToDevice, stream);
    (void)hipMemcpyAsync(out_update, all_span, (size_t)B_ * N_ * D_ * sizeof(float),
                         hipMemcpyDeviceToDevice, stream);
    scatter_kernel<<<BK_, 256, 0, stream>>>(ucur, prune, span_lengths, out_update);
}
